// HourglassDiffusionTransformer_24386824306810
// MI455X (gfx1250) — matrix operand order repaired and checked by reading, not compiled, not run on hardware
//
#include <hip/hip_runtime.h>

// ---------------------------------------------------------------------------
// Types for CDNA5 WMMA (wave32, 16x16x32 bf16 -> f32)
// ---------------------------------------------------------------------------
typedef __attribute__((ext_vector_type(16))) __bf16 v16bf;
typedef __attribute__((ext_vector_type(8)))  float  v8f;

__device__ __forceinline__ __bf16 f2bf(float f) {
  unsigned int u = __builtin_bit_cast(unsigned int, f);
  unsigned int r = (u + 0x7FFFu + ((u >> 16) & 1u)) >> 16;
  unsigned short s = (unsigned short)r;
  return __builtin_bit_cast(__bf16, s);
}

__device__ __forceinline__ int imin(int a, int b) { return a < b ? a : b; }
__device__ __forceinline__ int imax(int a, int b) { return a > b ? a : b; }

// ---------------------------------------------------------------------------
// Generic GEMM: C[M,N] = A[M,K] @ W[K,N]   (A row stride lda, C row stride ldc)
// fp32 in global, bf16 tiles in LDS, f32 accumulation via v_wmma_f32_16x16x32_bf16
// Block: 128 threads (4 waves). Block tile 128x64, K-step 32.
// Wave w computes rows [32w,32w+32) x 64 cols = 2x4 WMMA accumulators;
// each B fragment is reused for 2 WMMAs (DS ops per WMMA ~3).
// Requirements satisfied by every call here: K % 32 == 0, N % 64 == 0,
// A/W base and lda/N strides 16B-aligned.
// ---------------------------------------------------------------------------
#define BM 128
#define BN 64
#define BK 32
#define BKP 36   // LDS row pitch in halfwords: 72B = 8B-aligned rows, conflict-free

__global__ __launch_bounds__(128) void k_gemm(const float* __restrict__ A,
                                              const float* __restrict__ W,
                                              float* __restrict__ C,
                                              int M, int N, int K, int lda, int ldc)
{
  __shared__ __bf16 As[BM][BKP];
  __shared__ __bf16 Bs[BN][BKP];

  const int tid  = (int)threadIdx.x;
  const int lane = tid & 31;
  const int wave = tid >> 5;
  const int mblk = (int)blockIdx.y * BM;
  const int nblk = (int)blockIdx.x * BN;

  v8f acc[2][4];
#pragma unroll
  for (int h = 0; h < 2; ++h)
#pragma unroll
    for (int j = 0; j < 4; ++j)
#pragma unroll
      for (int r = 0; r < 8; ++r) acc[h][j][r] = 0.f;

  const int mrow0 = wave * 32 + (lane & 15);  // first A-frag row in block tile
  const int abase = (lane < 16) ? 0 : 16;     // contiguous k order: k = abase + e
  const int bbase = (lane < 16) ? 0 : 16;     // contiguous k order: k = bbase + e (same as A)
  const int ncol  = lane & 15;

  for (int k0 = 0; k0 < K; k0 += BK) {
    // --- stage A tile (128x32) with float4 loads, fp32 -> bf16 ---
#pragma unroll
    for (int i = tid; i < BM * BK / 4; i += 128) {   // 8 iterations
      int r  = i >> 3;           // 8 float4 per 32-wide row
      int c4 = (i & 7) * 4;
      int gm = mblk + r;
      float4 v = make_float4(0.f, 0.f, 0.f, 0.f);
      if (gm < M) v = *(const float4*)&A[(size_t)gm * lda + (k0 + c4)];
      As[r][c4 + 0] = f2bf(v.x);
      As[r][c4 + 1] = f2bf(v.y);
      As[r][c4 + 2] = f2bf(v.z);
      As[r][c4 + 3] = f2bf(v.w);
    }
    // --- stage B tile transposed to [n][k] with float4 loads ---
#pragma unroll
    for (int i = tid; i < BN * BK / 4; i += 128) {   // 4 iterations
      int kk = i >> 4;           // 16 float4 per 64-wide k-row
      int c4 = (i & 15) * 4;
      int gn = nblk + c4;
      float4 v = make_float4(0.f, 0.f, 0.f, 0.f);
      if (gn < N) v = *(const float4*)&W[(size_t)(k0 + kk) * N + gn];
      Bs[c4 + 0][kk] = f2bf(v.x);
      Bs[c4 + 1][kk] = f2bf(v.y);
      Bs[c4 + 2][kk] = f2bf(v.z);
      Bs[c4 + 3][kk] = f2bf(v.w);
    }
    // prefetch next A tile rows (speculative)
    if (k0 + BK < K) {
      int pr = mblk + tid;
      if (pr < M) __builtin_prefetch(&A[(size_t)pr * lda + (k0 + BK)], 0, 1);
    }
    __syncthreads();

    // --- build the two A fragments (contiguous k order, k = abase + e) ---
    v16bf af0, af1;
#pragma unroll
    for (int e = 0; e < 16; ++e) {
      int kk = abase + e;
      af0[e] = As[mrow0][kk];
      af1[e] = As[mrow0 + 16][kk];
    }
    // --- 4 column tiles; each B fragment feeds 2 WMMAs ---
#pragma unroll
    for (int j = 0; j < 4; ++j) {
      v16bf bfr;
#pragma unroll
      for (int e = 0; e < 16; ++e) bfr[e] = Bs[j * 16 + ncol][bbase + e];
      acc[0][j] = __builtin_amdgcn_wmma_f32_16x16x32_bf16(
          false, af0, false, bfr, (short)0, acc[0][j], false, false);
      acc[1][j] = __builtin_amdgcn_wmma_f32_16x16x32_bf16(
          false, af1, false, bfr, (short)0, acc[1][j], false, false);
    }
    __syncthreads();
  }

  // --- store C (ISA f32 16x16 C/D layout) ---
  const int rofs = (lane < 16) ? 0 : 8;
#pragma unroll
  for (int h = 0; h < 2; ++h)
#pragma unroll
    for (int j = 0; j < 4; ++j)
#pragma unroll
      for (int r = 0; r < 8; ++r) {
        int gm = mblk + wave * 32 + h * 16 + rofs + r;
        int gn = nblk + j * 16 + ncol;
        if (gm < M && gn < N) C[(size_t)gm * ldc + gn] = acc[h][j][r];
      }
}

// ---------------------------------------------------------------------------
// Flash-style attention (head dim 64). Q/K/V layout [B,H,S,64].
// Output merged heads: [B,S,H*64].  win>0: sliding window (query window qb
// attends keys in [(qb-1)*win, (qb+2)*win) clamped) == reference local_attn.
// win==0: full attention. Block = 128 threads, one query each.
// ---------------------------------------------------------------------------
__global__ __launch_bounds__(128) void k_attn(const float* __restrict__ Qp,
                                              const float* __restrict__ Kp,
                                              const float* __restrict__ Vp,
                                              float* __restrict__ Op,
                                              int S, int H, int win)
{
  __shared__ float Ks[64][65];
  __shared__ float Vs[64][65];
  const int b = (int)blockIdx.z, h = (int)blockIdx.y, qb = (int)blockIdx.x;
  const int tid = (int)threadIdx.x;
  const int qi = qb * 128 + tid;
  const size_t base = ((size_t)b * H + h) * S;
  const bool act = qi < S;

  float q[64];
#pragma unroll
  for (int d = 0; d < 64; ++d) q[d] = 0.f;
  if (act) {
#pragma unroll
    for (int d = 0; d < 64; ++d) q[d] = Qp[(base + qi) * 64 + d];
  }

  int k0 = 0, k1 = S;
  if (win > 0) { k0 = imax(0, (qb - 1) * win); k1 = imin(S, (qb + 2) * win); }

  float m = -3.0e38f, l = 0.f;
  float acc[64];
#pragma unroll
  for (int d = 0; d < 64; ++d) acc[d] = 0.f;

  for (int kc = k0; kc < k1; kc += 64) {
    int nk = imin(64, k1 - kc);
    for (int i = tid; i < 64 * 64; i += 128) {
      int r = i >> 6, c = i & 63;
      float kv = 0.f, vv = 0.f;
      if (r < nk) {
        size_t g = (base + kc + r) * 64 + c;
        kv = Kp[g]; vv = Vp[g];
      }
      Ks[r][c] = kv; Vs[r][c] = vv;
    }
    __syncthreads();
    if (act) {
      for (int j = 0; j < nk; ++j) {
        float s = 0.f;
#pragma unroll
        for (int d = 0; d < 64; ++d) s += q[d] * Ks[j][d];
        s *= 0.125f;  // 64^-0.5
        float mn = fmaxf(m, s);
        float cf = __expf(m - mn);
        float p  = __expf(s - mn);
        l = l * cf + p;
#pragma unroll
        for (int d = 0; d < 64; ++d) acc[d] = acc[d] * cf + p * Vs[j][d];
        m = mn;
      }
    }
    __syncthreads();
  }

  if (act) {
    float inv = 1.f / l;
    size_t o = ((size_t)b * S + qi) * ((size_t)H * 64) + (size_t)h * 64;
#pragma unroll
    for (int d = 0; d < 64; ++d) Op[o + d] = acc[d] * inv;
  }
}

// ---------------------------------------------------------------------------
// QKV split + RoPE (contiguous-half rotation, head dim 64).
// qkv [B,S,3C] -> Q,K,V [B,H,S,64]; rope applied to Q,K.
// ---------------------------------------------------------------------------
__global__ void k_qkv_rope(const float* __restrict__ qkv,
                           float* __restrict__ Qo, float* __restrict__ Ko,
                           float* __restrict__ Vo,
                           int S, int C, int H, size_t n)
{
  size_t idx = (size_t)blockIdx.x * 256 + threadIdx.x;
  if (idx >= n) return;
  int d = (int)(idx % 64);
  size_t rest = idx / 64;
  int hh = (int)(rest % H); rest /= H;
  int s  = (int)(rest % S);
  int b  = (int)(rest / S);

  size_t in_row = ((size_t)b * S + s) * (size_t)(3 * C);
  int ch = hh * 64 + d;
  float qv = qkv[in_row + ch];
  float kv = qkv[in_row + C + ch];
  float vv = qkv[in_row + 2 * C + ch];

  int i = d & 31;
  float inv = __expf(-(float)i * 0.2878231366f);  // ln(10000)/32
  float ang = (float)s * inv;
  float cs = __cosf(ang), sn = __sinf(ang);

  int partner = (d < 32) ? ch + 32 : ch - 32;
  float qp = qkv[in_row + partner];
  float kp = qkv[in_row + C + partner];
  float qo, ko;
  if (d < 32) { qo = qv * cs - qp * sn; ko = kv * cs - kp * sn; }
  else        { qo = qv * cs + qp * sn; ko = kv * cs + kp * sn; }

  size_t o = (((size_t)b * H + hh) * S + s) * 64 + d;
  Qo[o] = qo; Ko[o] = ko; Vo[o] = vv;
}

// ---------------------------------------------------------------------------
// Norm kernels (one block of 256 threads per row)
// ---------------------------------------------------------------------------
__global__ __launch_bounds__(256) void k_rms_vec(const float* __restrict__ X,
                                                 const float* __restrict__ sc,
                                                 float* __restrict__ Y, int C)
{
  __shared__ float red[256];
  const size_t row = blockIdx.x;
  const float* x = X + row * (size_t)C;
  float ss = 0.f;
  for (int c = (int)threadIdx.x; c < C; c += 256) { float v = x[c]; ss += v * v; }
  red[threadIdx.x] = ss; __syncthreads();
  for (int st = 128; st > 0; st >>= 1) {
    if ((int)threadIdx.x < st) red[threadIdx.x] += red[threadIdx.x + st];
    __syncthreads();
  }
  float r = rsqrtf(red[0] / (float)C + 1e-6f);
  float* y = Y + row * (size_t)C;
  for (int c = (int)threadIdx.x; c < C; c += 256) y[c] = x[c] * sc[c] * r;
}

__global__ __launch_bounds__(256) void k_ada_rms(const float* __restrict__ X,
                                                 const float* __restrict__ sA,
                                                 float* __restrict__ Y, int S, int C)
{
  __shared__ float red[256];
  const size_t row = blockIdx.x;
  const int b = (int)(row / (size_t)S);
  const float* x = X + row * (size_t)C;
  float ss = 0.f;
  for (int c = (int)threadIdx.x; c < C; c += 256) { float v = x[c]; ss += v * v; }
  red[threadIdx.x] = ss; __syncthreads();
  for (int st = 128; st > 0; st >>= 1) {
    if ((int)threadIdx.x < st) red[threadIdx.x] += red[threadIdx.x + st];
    __syncthreads();
  }
  float r = rsqrtf(red[0] / (float)C + 1e-6f);
  float* y = Y + row * (size_t)C;
  const float* s = sA + (size_t)b * C;
  for (int c = (int)threadIdx.x; c < C; c += 256) y[c] = x[c] * (1.f + s[c]) * r;
}

__global__ __launch_bounds__(256) void k_layernorm(const float* __restrict__ X,
                                                   const float* __restrict__ g,
                                                   const float* __restrict__ bb,
                                                   float* __restrict__ Y, int C)
{
  __shared__ float red[256];
  __shared__ float red2[256];
  const size_t row = blockIdx.x;
  const float* x = X + row * (size_t)C;
  float s1 = 0.f, s2 = 0.f;
  for (int c = (int)threadIdx.x; c < C; c += 256) { float v = x[c]; s1 += v; s2 += v * v; }
  red[threadIdx.x] = s1; red2[threadIdx.x] = s2; __syncthreads();
  for (int st = 128; st > 0; st >>= 1) {
    if ((int)threadIdx.x < st) {
      red[threadIdx.x]  += red[threadIdx.x + st];
      red2[threadIdx.x] += red2[threadIdx.x + st];
    }
    __syncthreads();
  }
  float mean = red[0] / (float)C;
  float var  = red2[0] / (float)C - mean * mean;
  float r = rsqrtf(var + 1e-5f);
  float* y = Y + row * (size_t)C;
  for (int c = (int)threadIdx.x; c < C; c += 256)
    y[c] = (x[c] - mean) * r * g[c] + bb[c];
}

// ---------------------------------------------------------------------------
// Elementwise kernels
// ---------------------------------------------------------------------------
__global__ void k_geglu(float* __restrict__ U, int F, size_t n)  // in-place gate
{
  size_t idx = (size_t)blockIdx.x * 256 + threadIdx.x;
  if (idx >= n) return;
  size_t row = idx / (size_t)F;
  int f = (int)(idx % (size_t)F);
  size_t base = row * (size_t)(2 * F);
  float a = U[base + f], g = U[base + F + f];
  float t = 0.7978845608f * (g + 0.044715f * g * g * g);
  U[base + f] = a * 0.5f * g * (1.f + tanhf(t));
}

__global__ void k_swiglu(float* __restrict__ U, int F, size_t n)  // in-place gate
{
  size_t idx = (size_t)blockIdx.x * 256 + threadIdx.x;
  if (idx >= n) return;
  size_t row = idx / (size_t)F;
  int f = (int)(idx % (size_t)F);
  size_t base = row * (size_t)(2 * F);
  float a = U[base + f], g = U[base + F + f];
  float sig = 1.f / (1.f + __expf(-g));
  U[base + f] = a * g * sig;
}

__global__ void k_add(float* __restrict__ Y, const float* __restrict__ D, size_t n)
{
  size_t idx = (size_t)blockIdx.x * 256 + threadIdx.x;
  if (idx < n) Y[idx] += D[idx];
}

__global__ void k_lerp(const float* __restrict__ skip, const float* __restrict__ xn,
                       const float* __restrict__ fac, float* __restrict__ Y, size_t n)
{
  size_t idx = (size_t)blockIdx.x * 256 + threadIdx.x;
  if (idx < n) { float f = fac[0]; Y[idx] = skip[idx] + f * (xn[idx] - skip[idx]); }
}

// token_merge (r=2): out[b,n,2c+rr] = x[b,2n+rr,c]    x:[B,S,C] out:[B,S/2,2C]
__global__ void k_merge2(const float* __restrict__ x, float* __restrict__ out,
                         int S, int C, size_t n)
{
  size_t idx = (size_t)blockIdx.x * 256 + threadIdx.x;
  if (idx >= n) return;
  int twoC = 2 * C;
  int j = (int)(idx % (size_t)twoC);
  size_t rest = idx / (size_t)twoC;
  int half = S / 2;
  int nn = (int)(rest % (size_t)half);
  int b  = (int)(rest / (size_t)half);
  int c = j >> 1, rr = j & 1;
  out[idx] = x[((size_t)b * S + 2 * nn + rr) * (size_t)C + c];
}

// token_split (r=2): out[b,2n+rr,c] = y[b,n,2c+rr]    y:[B,S/2,2C] out:[B,S,C]
__global__ void k_split2(const float* __restrict__ y, float* __restrict__ out,
                         int S, int C, size_t n)
{
  size_t idx = (size_t)blockIdx.x * 256 + threadIdx.x;
  if (idx >= n) return;
  int c = (int)(idx % (size_t)C);
  size_t rest = idx / (size_t)C;
  int s = (int)(rest % (size_t)S);
  int b = (int)(rest / (size_t)S);
  int nn = s >> 1, rr = s & 1;
  out[idx] = y[((size_t)b * (S / 2) + nn) * (size_t)(2 * C) + 2 * c + rr];
}

// patchify: x[B,64,4096] -> out[B,2048,128], out[b,n,2c+rr] = x[b,c,2n+rr]
__global__ void k_patchify(const float* __restrict__ x, float* __restrict__ out, size_t n)
{
  size_t idx = (size_t)blockIdx.x * 256 + threadIdx.x;
  if (idx >= n) return;
  int j = (int)(idx % 128);
  size_t rest = idx / 128;
  int nn = (int)(rest % 2048);
  int b  = (int)(rest / 2048);
  int c = j >> 1, rr = j & 1;
  out[idx] = x[((size_t)b * 64 + c) * 4096 + 2 * nn + rr];
}

// unpatchify: y[B,2048,128] -> out[B,64,4096], out[b,c,2n+rr] = y[b,n,2c+rr]
__global__ void k_unpatchify(const float* __restrict__ y, float* __restrict__ out, size_t n)
{
  size_t idx = (size_t)blockIdx.x * 256 + threadIdx.x;
  if (idx >= n) return;
  int t = (int)(idx % 4096);
  size_t rest = idx / 4096;
  int c = (int)(rest % 64);
  int b = (int)(rest / 64);
  int nn = t >> 1, rr = t & 1;
  out[idx] = y[((size_t)b * 2048 + nn) * 128 + 2 * c + rr];
}

// concat mapping token: out[b,0,:]=pre[b,:], out[b,1+s,:]=hin[b,s,:]
__global__ void k_concat_tok(const float* __restrict__ pre, const float* __restrict__ hin,
                             float* __restrict__ out, int Sin, int C, size_t n)
{
  size_t idx = (size_t)blockIdx.x * 256 + threadIdx.x;
  if (idx >= n) return;
  int c = (int)(idx % (size_t)C);
  size_t rest = idx / (size_t)C;
  int s = (int)(rest % (size_t)(Sin + 1));
  int b = (int)(rest / (size_t)(Sin + 1));
  out[idx] = (s == 0) ? pre[(size_t)b * C + c]
                      : hin[((size_t)b * Sin + (s - 1)) * (size_t)C + c];
}

// drop mapping token: out[b,s,:] = in[b,s+1,:]
__global__ void k_drop_tok(const float* __restrict__ in, float* __restrict__ out,
                           int Sout, int C, size_t n)
{
  size_t idx = (size_t)blockIdx.x * 256 + threadIdx.x;
  if (idx >= n) return;
  int c = (int)(idx % (size_t)C);
  size_t rest = idx / (size_t)C;
  int s = (int)(rest % (size_t)Sout);
  int b = (int)(rest / (size_t)Sout);
  out[idx] = in[((size_t)b * (Sout + 1) + s + 1) * (size_t)C + c];
}

// Fourier time embedding: out[b,j]=cos(2pi t[b] tw[j]), out[b,512+j]=sin(...)
__global__ void k_timeemb(const float* __restrict__ t, const float* __restrict__ tw,
                          float* __restrict__ out, size_t n)
{
  size_t idx = (size_t)blockIdx.x * 256 + threadIdx.x;
  if (idx >= n) return;
  int j = (int)(idx % 512);
  int b = (int)(idx / 512);
  float f = 6.28318530718f * t[b] * tw[j];
  out[(size_t)b * 1024 + j]       = cosf(f);
  out[(size_t)b * 1024 + 512 + j] = sinf(f);
}

// ---------------------------------------------------------------------------
// Host orchestration
// ---------------------------------------------------------------------------
static inline dim3 G1(size_t n) { return dim3((unsigned)((n + 255) / 256)); }

extern "C" void kernel_launch(void* const* d_in, const int* in_sizes, int n_in,
                              void* d_out, int out_size, void* d_ws, size_t ws_size,
                              hipStream_t stream) {
  (void)in_sizes; (void)n_in; (void)out_size;

  const float* in_x        = (const float*)d_in[0];
  const float* in_t        = (const float*)d_in[1];
  const float* time_w      = (const float*)d_in[2];
  const float* time_in_w   = (const float*)d_in[3];
  const float* map_in_s    = (const float*)d_in[4];
  const float* map_norm_s  = (const float*)d_in[5];
  const float* map_up_w    = (const float*)d_in[6];
  const float* map_down_w  = (const float*)d_in[7];
  const float* map_out_s   = (const float*)d_in[8];
  const float* patch_in_w  = (const float*)d_in[9];
  const float* dn_anorm_w  = (const float*)d_in[10];
  const float* dn_qkv_w    = (const float*)d_in[11];
  const float* dn_o_w      = (const float*)d_in[12];
  const float* dn_fnorm_w  = (const float*)d_in[13];
  const float* dn_ff1_w    = (const float*)d_in[14];
  const float* dn_ff2_w    = (const float*)d_in[15];
  const float* merge_w     = (const float*)d_in[16];
  const float* mid_map_w   = (const float*)d_in[17];
  const float* mid_pin_w   = (const float*)d_in[18];
  const float* mid_pout_w  = (const float*)d_in[19];
  const float* mid_ln1_g   = (const float*)d_in[20];
  const float* mid_ln1_b   = (const float*)d_in[21];
  const float* mid_qkv_w   = (const float*)d_in[22];
  const float* mid_o_w     = (const float*)d_in[23];
  const float* mid_ln2_g   = (const float*)d_in[24];
  const float* mid_ln2_b   = (const float*)d_in[25];
  const float* mid_ff1_w   = (const float*)d_in[26];
  const float* mid_ff2_w   = (const float*)d_in[27];
  const float* mid_fn_g    = (const float*)d_in[28];
  const float* mid_fn_b    = (const float*)d_in[29];
  const float* split_w     = (const float*)d_in[30];
  const float* split_fac   = (const float*)d_in[31];
  const float* up_anorm_w  = (const float*)d_in[32];
  const float* up_qkv_w    = (const float*)d_in[33];
  const float* up_o_w      = (const float*)d_in[34];
  const float* up_fnorm_w  = (const float*)d_in[35];
  const float* up_ff1_w    = (const float*)d_in[36];
  const float* up_ff2_w    = (const float*)d_in[37];
  const float* out_norm_s  = (const float*)d_in[38];
  const float* patch_out_w = (const float*)d_in[39];

  const int Bb = 2, CIO = 64, T = 4096;
  const int S0 = 2048, S1 = 1024, SM = 1025;
  const int W0 = 512, W1 = 1024, MAPD = 1024;
  const int H0 = 8, H1 = 16, WIN = 128;

  // ---- workspace layout (floats) ----
  float* w = (float*)d_ws;
  size_t off = 0;
  const size_t CAP1 = (size_t)2 * 1025 * 1024;  // covers every [B,S,C] tensor here
  float* buf_h    = w + off; off += CAP1;
  float* buf_n    = w + off; off += CAP1;
  float* buf_att  = w + off; off += CAP1;
  float* buf_tmp  = w + off; off += CAP1;
  float* buf_skip = w + off; off += CAP1;
  float* buf_q    = w + off; off += CAP1;
  float* buf_k    = w + off; off += CAP1;
  float* buf_v    = w + off; off += CAP1;
  float* buf_qkv  = w + off; off += (size_t)2 * 1025 * 3072;
  float* buf_ff   = w + off; off += (size_t)2 * 1025 * 8192;
  float* misc     = w + off; off += 32768;
  float* buf_m = buf_qkv;  // token merge/split scratch aliases idle QKV buffer
  if (off * sizeof(float) > ws_size) return;

  float* temb_in = misc;           // [B,1024]
  float* temb    = misc + 2048;    // [B,1024]
  float* xmap    = misc + 4096;    // [B,1024]
  float* hn_map  = misc + 6144;    // [B,1024]
  float* map_u   = misc + 8192;    // [B,2048]
  float* map_d   = misc + 12288;   // [B,1024]
  float* cond    = misc + 14336;   // [B,1024]
  float* sA      = misc + 16384;   // [B,512]
  float* pre     = misc + 17408;   // [B,1024]

  auto gemm = [&](const float* A, const float* Wp, float* Cp,
                  int M, int N, int K, int lda, int ldc) {
    dim3 grid((unsigned)((N + BN - 1) / BN), (unsigned)((M + BM - 1) / BM));
    k_gemm<<<grid, 128, 0, stream>>>(A, Wp, Cp, M, N, K, lda, ldc);
  };

  // ---- patchify + input projection ----
  {
    size_t n = (size_t)Bb * S0 * (CIO * 2);
    k_patchify<<<G1(n), 256, 0, stream>>>(in_x, buf_m, n);
  }
  gemm(buf_m, patch_in_w, buf_h, Bb * S0, W0, CIO * 2, CIO * 2, W0);

  // ---- time embedding + mapping network -> cond [B,1024] ----
  k_timeemb<<<G1((size_t)Bb * 512), 256, 0, stream>>>(in_t, time_w, temb_in, (size_t)Bb * 512);
  gemm(temb_in, time_in_w, temb, Bb, MAPD, MAPD, MAPD, MAPD);
  k_rms_vec<<<dim3(Bb), 256, 0, stream>>>(temb, map_in_s, xmap, MAPD);
  for (int l = 0; l < 2; ++l) {
    k_rms_vec<<<dim3(Bb), 256, 0, stream>>>(xmap, map_norm_s + (size_t)l * MAPD, hn_map, MAPD);
    gemm(hn_map, map_up_w + (size_t)l * MAPD * 2048, map_u, Bb, 2048, MAPD, MAPD, 2048);
    k_geglu<<<G1((size_t)Bb * 1024), 256, 0, stream>>>(map_u, 1024, (size_t)Bb * 1024);
    gemm(map_u, map_down_w + (size_t)l * 1024 * MAPD, map_d, Bb, MAPD, 1024, 2048, MAPD);
    k_add<<<G1((size_t)Bb * MAPD), 256, 0, stream>>>(xmap, map_d, (size_t)Bb * MAPD);
  }
  k_rms_vec<<<dim3(Bb), 256, 0, stream>>>(xmap, map_out_s, cond, MAPD);

  // ---- local level runner (2 layers, GEGLU FF, local attention) ----
  auto run_local = [&](const float* anorm, const float* qkvw, const float* ow,
                       const float* fnorm, const float* ff1w, const float* ff2w) {
    const int S = S0, C = W0, H = H0;
    const size_t rows = (size_t)Bb * S;
    for (int l = 0; l < 2; ++l) {
      gemm(cond, anorm + (size_t)l * MAPD * C, sA, Bb, C, MAPD, MAPD, C);
      k_ada_rms<<<dim3((unsigned)rows), 256, 0, stream>>>(buf_h, sA, buf_n, S, C);
      gemm(buf_n, qkvw + (size_t)l * C * 3 * C, buf_qkv, (int)rows, 3 * C, C, C, 3 * C);
      {
        size_t n = rows * (size_t)C;
        k_qkv_rope<<<G1(n), 256, 0, stream>>>(buf_qkv, buf_q, buf_k, buf_v, S, C, H, n);
      }
      k_attn<<<dim3((unsigned)(S / WIN), (unsigned)H, (unsigned)Bb), 128, 0, stream>>>(
          buf_q, buf_k, buf_v, buf_att, S, H, WIN);
      gemm(buf_att, ow + (size_t)l * C * C, buf_tmp, (int)rows, C, C, C, C);
      k_add<<<G1(rows * C), 256, 0, stream>>>(buf_h, buf_tmp, rows * (size_t)C);
      gemm(cond, fnorm + (size_t)l * MAPD * C, sA, Bb, C, MAPD, MAPD, C);
      k_ada_rms<<<dim3((unsigned)rows), 256, 0, stream>>>(buf_h, sA, buf_n, S, C);
      gemm(buf_n, ff1w + (size_t)l * C * 4 * C, buf_ff, (int)rows, 4 * C, C, C, 4 * C);
      k_geglu<<<G1(rows * (size_t)(2 * C)), 256, 0, stream>>>(buf_ff, 2 * C, rows * (size_t)(2 * C));
      gemm(buf_ff, ff2w + (size_t)l * 2 * C * C, buf_tmp, (int)rows, C, 2 * C, 4 * C, C);
      k_add<<<G1(rows * C), 256, 0, stream>>>(buf_h, buf_tmp, rows * (size_t)C);
    }
  };

  // ---- down level ----
  run_local(dn_anorm_w, dn_qkv_w, dn_o_w, dn_fnorm_w, dn_ff1_w, dn_ff2_w);

  // ---- skip ----
  hipMemcpyAsync(buf_skip, buf_h, (size_t)Bb * S0 * W0 * sizeof(float),
                 hipMemcpyDeviceToDevice, stream);

  // ---- merge + pin + prepend mapping token ----
  {
    size_t n = (size_t)Bb * S1 * (2 * W0);
    k_merge2<<<G1(n), 256, 0, stream>>>(buf_h, buf_m, S0, W0, n);
  }
  gemm(buf_m, merge_w, buf_tmp, Bb * S1, W1, 2 * W0, 2 * W0, W1);   // h1
  gemm(buf_tmp, mid_pin_w, buf_n, Bb * S1, W1, W1, W1, W1);         // pin
  gemm(cond, mid_map_w, pre, Bb, W1, MAPD, MAPD, W1);
  {
    size_t n = (size_t)Bb * SM * W1;
    k_concat_tok<<<G1(n), 256, 0, stream>>>(pre, buf_n, buf_h, S1, W1, n);
  }

  // ---- mid level: 4 layers, full attention, SwiGLU FF ----
  const size_t rowsM = (size_t)Bb * SM;
  for (int l = 0; l < 4; ++l) {
    k_layernorm<<<dim3((unsigned)rowsM), 256, 0, stream>>>(
        buf_h, mid_ln1_g + (size_t)l * W1, mid_ln1_b + (size_t)l * W1, buf_n, W1);
    gemm(buf_n, mid_qkv_w + (size_t)l * W1 * 3 * W1, buf_qkv, (int)rowsM, 3 * W1, W1, W1, 3 * W1);
    {
      size_t n = rowsM * (size_t)W1;
      k_qkv_rope<<<G1(n), 256, 0, stream>>>(buf_qkv, buf_q, buf_k, buf_v, SM, W1, H1, n);
    }
    k_attn<<<dim3((unsigned)((SM + 127) / 128), (unsigned)H1, (unsigned)Bb), 128, 0, stream>>>(
        buf_q, buf_k, buf_v, buf_att, SM, H1, 0);
    gemm(buf_att, mid_o_w + (size_t)l * W1 * W1, buf_tmp, (int)rowsM, W1, W1, W1, W1);
    k_add<<<G1(rowsM * W1), 256, 0, stream>>>(buf_h, buf_tmp, rowsM * (size_t)W1);
    k_layernorm<<<dim3((unsigned)rowsM), 256, 0, stream>>>(
        buf_h, mid_ln2_g + (size_t)l * W1, mid_ln2_b + (size_t)l * W1, buf_n, W1);
    gemm(buf_n, mid_ff1_w + (size_t)l * W1 * 8 * W1, buf_ff, (int)rowsM, 8 * W1, W1, W1, 8 * W1);
    k_swiglu<<<G1(rowsM * (size_t)(4 * W1)), 256, 0, stream>>>(buf_ff, 4 * W1, rowsM * (size_t)(4 * W1));
    gemm(buf_ff, mid_ff2_w + (size_t)l * 4 * W1 * W1, buf_tmp, (int)rowsM, W1, 4 * W1, 8 * W1, W1);
    k_add<<<G1(rowsM * W1), 256, 0, stream>>>(buf_h, buf_tmp, rowsM * (size_t)W1);
  }
  k_layernorm<<<dim3((unsigned)rowsM), 256, 0, stream>>>(buf_h, mid_fn_g, mid_fn_b, buf_n, W1);

  // ---- pout, drop token, split, lerp skip fusion ----
  gemm(buf_n, mid_pout_w, buf_tmp, (int)rowsM, W1, W1, W1, W1);
  {
    size_t n = (size_t)Bb * S1 * W1;
    k_drop_tok<<<G1(n), 256, 0, stream>>>(buf_tmp, buf_m, S1, W1, n);
  }
  gemm(buf_m, split_w, buf_tmp, Bb * S1, 2 * W0, W1, W1, 2 * W0);
  {
    size_t n = (size_t)Bb * S0 * W0;
    k_split2<<<G1(n), 256, 0, stream>>>(buf_tmp, buf_n, S0, W0, n);
    k_lerp<<<G1(n), 256, 0, stream>>>(buf_skip, buf_n, split_fac, buf_h, n);
  }

  // ---- up level ----
  run_local(up_anorm_w, up_qkv_w, up_o_w, up_fnorm_w, up_ff1_w, up_ff2_w);

  // ---- output head ----
  k_rms_vec<<<dim3((unsigned)((size_t)Bb * S0)), 256, 0, stream>>>(buf_h, out_norm_s, buf_n, W0);
  gemm(buf_n, patch_out_w, buf_tmp, Bb * S0, CIO * 2, W0, W0, CIO * 2);
  {
    size_t n = (size_t)Bb * CIO * T;
    k_unpatchify<<<G1(n), 256, 0, stream>>>(buf_tmp, (float*)d_out, n);
  }
}
